// LSTMCell_78159814853182
// MI455X (gfx1250) — matrix operand order repaired and checked by reading, not compiled, not run on hardware
//
#include <hip/hip_runtime.h>
#include <hip/hip_bf16.h>
#include <math.h>

typedef __attribute__((ext_vector_type(16))) __bf16 v16bf;
typedef __attribute__((ext_vector_type(8)))  float  v8f;

#define B_DIM 8192
#define IN_DIM 2048
#define H_DIM 2048
#define KD 2048      // row length of x/h and of both weight matrices
#define BK 32
#define BM 128
#define BN 64
#define ASTR 40      // padded LDS row stride (bf16 elems)
#define BSTR 40
#define KSTEPS 128   // (IN + H) / BK

union FragU { uint4 u[2]; v16bf v; };

__device__ __forceinline__ float sigmoidf_(float x) {
    return 1.0f / (1.0f + __expf(-x));
}

__device__ __forceinline__ void async_b128(uint32_t lds_off, const void* gptr) {
    asm volatile("global_load_async_to_lds_b128 %0, %1, off"
                 :: "v"(lds_off), "v"((uint64_t)(uintptr_t)gptr)
                 : "memory");
}

__global__ void cvt_f32_bf16(const float* __restrict__ src,
                             __bf16* __restrict__ dst, int n) {
    int i = blockIdx.x * blockDim.x + threadIdx.x;
    int stride = gridDim.x * blockDim.x;
    for (; i < n; i += stride) dst[i] = (__bf16)src[i];
}

__global__ __launch_bounds__(256)
void lstm_wmma_kernel(const __bf16* __restrict__ xb,
                      const __bf16* __restrict__ hb,
                      const __bf16* __restrict__ wihb,
                      const __bf16* __restrict__ whhb,
                      const float* __restrict__ bias,
                      const float* __restrict__ c_prev,
                      float* __restrict__ out) {
    __shared__ alignas(16) __bf16 As[2][BM * ASTR];       // 2 x 10.0 KB
    __shared__ alignas(16) __bf16 Bs[2][4][BN * BSTR];    // 2 x 20.0 KB

    const int tid  = threadIdx.x;
    const int wave = tid >> 5;
    const int lane = tid & 31;
    const int wm   = wave & 3;    // 4 row-slices of 32
    const int wn   = wave >> 2;   // 2 col-slices of 32
    const int r    = lane & 15;
    const int half = lane >> 4;

    const int m0 = blockIdx.x * BM;
    const int n0 = blockIdx.y * BN;

    // Issue one k-tile's global->LDS async loads (6 x b128 per thread)
    auto issue_tile = [&](int kt, int buf) {
        const bool second = kt >= (KSTEPS / 2);
        const int kk = (second ? (kt - KSTEPS / 2) : kt) * BK;
        const __bf16* Ag = second ? hb : xb;
        const __bf16* Wg = second ? whhb : wihb;
        // A tile: 128 rows x 32 cols; 2 chunks of 8 bf16 per thread
        {
            const int row = tid >> 1;
            const int c0  = (tid & 1) * 16;
            const __bf16* p = Ag + (size_t)(m0 + row) * KD + kk + c0;
            const uint32_t l = (uint32_t)(uintptr_t)&As[buf][row * ASTR + c0];
            async_b128(l,      p);
            async_b128(l + 16, p + 8);
        }
        // B tiles: 4 gates x 64 rows x 32 cols; 1 chunk per gate per thread
        #pragma unroll
        for (int g = 0; g < 4; ++g) {
            const int row = tid >> 2;
            const int c0  = (tid & 3) * 8;
            const __bf16* p = Wg + (size_t)(g * H_DIM + n0 + row) * KD + kk + c0;
            const uint32_t l = (uint32_t)(uintptr_t)&Bs[buf][g][row * BSTR + c0];
            async_b128(l, p);
        }
    };

    v8f acc[4][2][2] = {};   // [gate][mf][nf]

    issue_tile(0, 0);

    for (int kt = 0; kt < KSTEPS; ++kt) {
        // My async writes for tile kt are done; barrier makes everyone's visible
        asm volatile("s_wait_asynccnt 0x0" ::: "memory");
        __syncthreads();
        // Prefetch next tile into the other buffer (overlaps with compute below);
        // safe: all waves finished reading that buffer before the barrier above.
        if (kt + 1 < KSTEPS) issue_tile(kt + 1, (kt + 1) & 1);

        const __bf16* Ab = As[kt & 1];

        // A fragments: two 16x32 tiles (rows wm*32 .. wm*32+31)
        FragU af[2];
        #pragma unroll
        for (int mf = 0; mf < 2; ++mf) {
            const __bf16* ap = &Ab[(wm * 32 + mf * 16 + r) * ASTR + half * 8];
            af[mf].u[0] = *(const uint4*)ap;          // K = half*8 + 0..7
            af[mf].u[1] = *(const uint4*)(ap + 16);   // K = 16 + half*8 + 0..7
        }
        #pragma unroll
        for (int g = 0; g < 4; ++g) {
            const __bf16* Bb = Bs[kt & 1][g];
            #pragma unroll
            for (int nf = 0; nf < 2; ++nf) {
                FragU bf;
                const __bf16* bp = &Bb[(wn * 32 + nf * 16 + r) * BSTR + half * 8];
                bf.u[0] = *(const uint4*)bp;       // K = half*8 + 0..7
                bf.u[1] = *(const uint4*)(bp + 16); // K = 16 + half*8 + 0..7
                #pragma unroll
                for (int mf = 0; mf < 2; ++mf) {
                    acc[g][mf][nf] = __builtin_amdgcn_wmma_f32_16x16x32_bf16(
                        false, af[mf].v, false, bf.v, (short)0,
                        acc[g][mf][nf], false, false);
                }
            }
        }
    }

    // Fused LSTM epilogue: acc[0..3] = i,f,j,o pre-activations for (m, col)
    #pragma unroll
    for (int nf = 0; nf < 2; ++nf) {
        const int col = n0 + wn * 32 + nf * 16 + r;
        const float bi  = bias[col];
        const float bf_ = bias[H_DIM + col];
        const float bj  = bias[2 * H_DIM + col];
        const float bo  = bias[3 * H_DIM + col];
        #pragma unroll
        for (int mf = 0; mf < 2; ++mf) {
            #pragma unroll
            for (int e = 0; e < 8; ++e) {
                const int m = m0 + wm * 32 + mf * 16 + e + half * 8;
                const float i_ = sigmoidf_(acc[0][mf][nf][e] + bi);
                const float f_ = sigmoidf_(acc[1][mf][nf][e] + bf_);
                const float j_ = tanhf(acc[2][mf][nf][e] + bj);
                const float o_ = sigmoidf_(acc[3][mf][nf][e] + bo);
                const float cp = c_prev[(size_t)m * H_DIM + col];
                const float cc = f_ * cp + fminf(1.0f - f_, i_) * j_;
                const float hh = o_ * tanhf(cc);
                out[(size_t)m * H_DIM + col] = hh;
                out[(size_t)B_DIM * H_DIM + (size_t)m * H_DIM + col] = cc;
            }
        }
    }
}

extern "C" void kernel_launch(void* const* d_in, const int* in_sizes, int n_in,
                              void* d_out, int out_size, void* d_ws, size_t ws_size,
                              hipStream_t stream) {
    const float* x    = (const float*)d_in[0];
    const float* h_p  = (const float*)d_in[1];
    const float* c_p  = (const float*)d_in[2];
    const float* wih  = (const float*)d_in[3];
    const float* whh  = (const float*)d_in[4];
    const float* bias = (const float*)d_in[5];

    char* ws = (char*)d_ws;
    const size_t nx = (size_t)B_DIM * IN_DIM;
    const size_t nh = (size_t)B_DIM * H_DIM;
    const size_t nw = (size_t)4 * H_DIM * KD;
    __bf16* xb   = (__bf16*)(ws);
    __bf16* hb   = (__bf16*)(ws + nx * 2);
    __bf16* wihb = (__bf16*)(ws + nx * 2 + nh * 2);
    __bf16* whhb = (__bf16*)(ws + nx * 2 + nh * 2 + nw * 2);

    // One-pass fp32 -> bf16 conversion: hot GEMM data becomes 134 MB < 192 MB L2
    cvt_f32_bf16<<<2048, 256, 0, stream>>>(x,   xb,   (int)nx);
    cvt_f32_bf16<<<2048, 256, 0, stream>>>(h_p, hb,   (int)nh);
    cvt_f32_bf16<<<2048, 256, 0, stream>>>(wih, wihb, (int)nw);
    cvt_f32_bf16<<<2048, 256, 0, stream>>>(whh, whhb, (int)nw);

    dim3 grid(B_DIM / BM, H_DIM / BN);   // 64 x 32 blocks
    lstm_wmma_kernel<<<grid, 256, 0, stream>>>(xb, hb, wihb, whhb,
                                               bias, c_p, (float*)d_out);
}
